// custom_gnn_3831110828331
// MI455X (gfx1250) — compile-verified, matrix operand order checked by reading, not run on hardware
//
#include <hip/hip_runtime.h>
#include <hip/hip_bf16.h>
#include <math.h>

#define N_NODES 50000
#define N_EDGES 400000
#define IN_DIM  386
#define HID     512
#define OUT_DIM 384
#define LAYERS  4
#define HEADS   4
#define HD      128

typedef __attribute__((ext_vector_type(16))) _Float16 v16h;
typedef __attribute__((ext_vector_type(8)))  float    v8f;

// ---------------------------------------------------------------------------
// fp32 -> f16 bulk conversion (8 elems/thread, b128 in / b128 out)
// ---------------------------------------------------------------------------
__global__ __launch_bounds__(256) void cvt_f32_f16(
    const float* __restrict__ x, _Float16* __restrict__ y, size_t n)
{
  size_t i = ((size_t)blockIdx.x * blockDim.x + threadIdx.x) * 8;
  if (i + 8 <= n) {
    const float4* xp = (const float4*)(x + i);
    float4 a = xp[0], b = xp[1];
    union { _Float16 h[8]; uint4 u; } pk;
    pk.h[0] = (_Float16)a.x; pk.h[1] = (_Float16)a.y;
    pk.h[2] = (_Float16)a.z; pk.h[3] = (_Float16)a.w;
    pk.h[4] = (_Float16)b.x; pk.h[5] = (_Float16)b.y;
    pk.h[6] = (_Float16)b.z; pk.h[7] = (_Float16)b.w;
    *(uint4*)(y + i) = pk.u;
  } else {
    for (size_t j = i; j < n; ++j) y[j] = (_Float16)x[j];
  }
}

// ---------------------------------------------------------------------------
// GEMM: C[M,N] = A16[M,K] @ W16[N,K]^T + bias[N]   (f16 in, f32 out, WMMA)
// Block tile 128x64, BK=32, 256 threads = 8 waves (wave32), wave tile 32x32.
// Branch-free inner loop: row-clamped staging + split K tail.
// ---------------------------------------------------------------------------
#define BM 128
#define BN 64
#define BK 32
#define LDS_A 40   // padded f16 row stride (80B, 16B-aligned)
#define LDS_B 40

union FragU { v16h v; uint4 u[2]; };

template <bool AL16>
__global__ __launch_bounds__(256) void gemm_wmma_t(
    const _Float16* __restrict__ A, const _Float16* __restrict__ W,
    const float* __restrict__ bias, float* __restrict__ C,
    int M, int N, int K)
{
  __shared__ _Float16 As[BM * LDS_A];
  __shared__ _Float16 Bs[BN * LDS_B];

  const int tid  = threadIdx.x;
  const int lane = tid & 31;
  const int wave = tid >> 5;
  const int wm   = wave >> 1;      // 0..3 (M direction)
  const int wn   = wave & 1;       // 0..1 (N direction)
  const int lrow = lane & 15;
  const int kg   = lane >> 4;      // K half-group per ISA 16-bit A/B layout

  const int mBase = blockIdx.y * BM;
  const int nBase = blockIdx.x * BN;

  v8f acc[2][2] = {};

  // staging maps
  const int arow = tid >> 1;         // 0..127, 16-elem (32B) col segment
  const int acol = (tid & 1) * 16;
  const int brow = tid >> 2;         // 0..63, 8-elem (16B) col segment
  const int bcol = (tid & 3) * 8;

  // Row-clamped A pointer: staging loads are always legal; duplicated rows
  // only feed accumulator rows the epilogue never stores.
  int growA = mBase + arow; if (growA > M - 1) growA = M - 1;
  const _Float16* Arow = A + (size_t)growA * K;
  const _Float16* Wrow = W + (size_t)(nBase + brow) * K;  // N multiple of BN

  uint4* dsa = (uint4*)&As[arow * LDS_A + acol];
  uint4* dsb = (uint4*)&Bs[brow * LDS_B + bcol];

  // fragment LDS base pointers
  const _Float16* apf[2];
  const _Float16* bpf[2];
  #pragma unroll
  for (int mi = 0; mi < 2; ++mi)
    apf[mi] = &As[(wm * 32 + mi * 16 + lrow) * LDS_A + kg * 8];
  #pragma unroll
  for (int ni = 0; ni < 2; ++ni)
    bpf[ni] = &Bs[(wn * 32 + ni * 16 + lrow) * LDS_B + kg * 8];

  const int kFull = K & ~(BK - 1);

  for (int k0 = 0; k0 < kFull; k0 += BK) {
    // ---- stage A tile: unconditional 32B copy ----
    {
      const _Float16* Ar = Arow + k0 + acol;
      if (AL16) {
        const uint4* g = (const uint4*)Ar;
        uint4 x0 = g[0], x1 = g[1];
        dsa[0] = x0; dsa[1] = x1;
      } else {
        const uint* g = (const uint*)Ar;          // 4B-aligned (K even)
        union { uint w[8]; uint4 u[2]; } pk;
        #pragma unroll
        for (int i = 0; i < 8; ++i) pk.w[i] = g[i];
        dsa[0] = pk.u[0]; dsa[1] = pk.u[1];
      }
    }
    // ---- stage W tile: unconditional 16B copy ----
    {
      const _Float16* Wr = Wrow + k0 + bcol;
      if (AL16) {
        *dsb = *(const uint4*)Wr;
      } else {
        const uint* g = (const uint*)Wr;
        union { uint w[4]; uint4 u; } pk;
        #pragma unroll
        for (int i = 0; i < 4; ++i) pk.w[i] = g[i];
        *dsb = pk.u;
      }
    }
    __syncthreads();

    FragU fa[2], fb[2];
    #pragma unroll
    for (int mi = 0; mi < 2; ++mi) {
      fa[mi].u[0] = *(const uint4*)apf[mi];
      fa[mi].u[1] = *(const uint4*)(apf[mi] + 16);
    }
    #pragma unroll
    for (int ni = 0; ni < 2; ++ni) {
      fb[ni].u[0] = *(const uint4*)bpf[ni];
      fb[ni].u[1] = *(const uint4*)(bpf[ni] + 16);
    }
    #pragma unroll
    for (int mi = 0; mi < 2; ++mi)
      #pragma unroll
      for (int ni = 0; ni < 2; ++ni)
        acc[mi][ni] = __builtin_amdgcn_wmma_f32_16x16x32_f16(
            false, fa[mi].v, false, fb[ni].v, (short)0, acc[mi][ni], false, false);
    __syncthreads();
  }

  // ---- K tail (only for K % 32 != 0, e.g. the 386-wide input GEMM) ----
  if (kFull < K) {
    {
      union { _Float16 hh[16]; uint4 u[2]; } pk;
      #pragma unroll
      for (int i = 0; i < 16; ++i) {
        int kk = kFull + acol + i;
        pk.hh[i] = (kk < K) ? Arow[kk] : (_Float16)0.0f;
      }
      dsa[0] = pk.u[0]; dsa[1] = pk.u[1];
    }
    {
      union { _Float16 hh[8]; uint4 u; } pk;
      #pragma unroll
      for (int i = 0; i < 8; ++i) {
        int kk = kFull + bcol + i;
        pk.hh[i] = (kk < K) ? Wrow[kk] : (_Float16)0.0f;
      }
      *dsb = pk.u;
    }
    __syncthreads();

    FragU fa[2], fb[2];
    #pragma unroll
    for (int mi = 0; mi < 2; ++mi) {
      fa[mi].u[0] = *(const uint4*)apf[mi];
      fa[mi].u[1] = *(const uint4*)(apf[mi] + 16);
    }
    #pragma unroll
    for (int ni = 0; ni < 2; ++ni) {
      fb[ni].u[0] = *(const uint4*)bpf[ni];
      fb[ni].u[1] = *(const uint4*)(bpf[ni] + 16);
    }
    #pragma unroll
    for (int mi = 0; mi < 2; ++mi)
      #pragma unroll
      for (int ni = 0; ni < 2; ++ni)
        acc[mi][ni] = __builtin_amdgcn_wmma_f32_16x16x32_f16(
            false, fa[mi].v, false, fb[ni].v, (short)0, acc[mi][ni], false, false);
    __syncthreads();
  }

  // Epilogue: C/D layout -> m = r + kg*8, n = lrow within each 16x16 tile
  #pragma unroll
  for (int mi = 0; mi < 2; ++mi) {
    #pragma unroll
    for (int ni = 0; ni < 2; ++ni) {
      int gcol = nBase + wn * 32 + ni * 16 + lrow;
      float bc = bias[gcol];
      #pragma unroll
      for (int r = 0; r < 8; ++r) {
        int grow = mBase + wm * 32 + mi * 16 + kg * 8 + r;
        if (grow < M) C[(size_t)grow * N + gcol] = acc[mi][ni][r] + bc;
      }
    }
  }
}

static inline void launch_gemm(const _Float16* A, const _Float16* W,
                               const float* bias, float* C,
                               int M, int N, int K, hipStream_t stream)
{
  dim3 grid(N / BN, (M + BM - 1) / BM), blk(256);
  if ((K & 7) == 0)
    gemm_wmma_t<true><<<grid, blk, 0, stream>>>(A, W, bias, C, M, N, K);
  else
    gemm_wmma_t<false><<<grid, blk, 0, stream>>>(A, W, bias, C, M, N, K);
}

// ---------------------------------------------------------------------------
// Per-layer init: msg = 0, node max = -inf, denom = 0
// ---------------------------------------------------------------------------
__global__ __launch_bounds__(256) void fill_layer(float* msg, float* nmax, float* denom)
{
  size_t i = (size_t)blockIdx.x * blockDim.x + threadIdx.x;
  if (i < (size_t)N_NODES * HID) msg[i] = 0.0f;
  if (i < (size_t)N_NODES * HEADS) { nmax[i] = -3.0e38f; denom[i] = 0.0f; }
}

__device__ inline void atomicMaxFloat(float* addr, float val)
{
  unsigned int* ua = (unsigned int*)addr;
  unsigned int old = *ua;
  while (__uint_as_float(old) < val) {
    unsigned int assumed = old;
    old = atomicCAS(ua, assumed, __float_as_uint(val));
    if (old == assumed) break;
  }
}

// ---------------------------------------------------------------------------
// Pass A: wave per edge, per-head dot(q[dst], k[src]) / sqrt(HD); atomic max.
// ---------------------------------------------------------------------------
__global__ __launch_bounds__(256) void edge_logits(
    const float* __restrict__ q, const float* __restrict__ k,
    const int* __restrict__ src, const int* __restrict__ dst,
    float* __restrict__ elog, float* __restrict__ nmax, int E)
{
  int wid  = (int)(((size_t)blockIdx.x * blockDim.x + threadIdx.x) >> 5);
  int lane = threadIdx.x & 31;
  if (wid >= E) return;
  int s = src[wid], d = dst[wid];
  const float4* qp = (const float4*)(q + (size_t)d * HID);
  const float4* kp = (const float4*)(k + (size_t)s * HID);
  float acc = 0.0f;
  #pragma unroll
  for (int i = 0; i < 4; ++i) {      // lane covers cols [lane*16, lane*16+16)
    float4 a = qp[lane * 4 + i];
    float4 b = kp[lane * 4 + i];
    acc += a.x * b.x + a.y * b.y + a.z * b.z + a.w * b.w;
  }
  // one head = 128 cols = 8 lanes; reduce within lane-groups of 8
  acc += __shfl_xor(acc, 1, 32);
  acc += __shfl_xor(acc, 2, 32);
  acc += __shfl_xor(acc, 4, 32);
  if ((lane & 7) == 0) {
    int h = lane >> 3;
    float logit = acc * 0.08838834764831845f;   // 1/sqrt(128)
    elog[(size_t)wid * HEADS + h] = logit;
    atomicMaxFloat(&nmax[(size_t)d * HEADS + h], logit);
  }
}

// ---------------------------------------------------------------------------
// Pass B: thread per (edge,head): e = exp(logit - max[dst]); denom += e
// ---------------------------------------------------------------------------
__global__ __launch_bounds__(256) void edge_exp(
    float* __restrict__ elog, const float* __restrict__ nmax,
    float* __restrict__ denom, const int* __restrict__ dst, int E)
{
  size_t t = (size_t)blockIdx.x * blockDim.x + threadIdx.x;
  if (t >= (size_t)E * HEADS) return;
  int e = (int)(t >> 2), h = (int)(t & 3);
  int d = dst[e];
  float x = __expf(elog[t] - nmax[(size_t)d * HEADS + h]);
  elog[t] = x;
  atomicAdd(&denom[(size_t)d * HEADS + h], x);
}

// ---------------------------------------------------------------------------
// Pass C: wave per edge: msg[dst] += alpha_h * v[src]
// ---------------------------------------------------------------------------
__global__ __launch_bounds__(256) void edge_scatter(
    const float* __restrict__ v, const float* __restrict__ elog,
    const float* __restrict__ denom, const int* __restrict__ src,
    const int* __restrict__ dst, float* __restrict__ msg, int E)
{
  int wid  = (int)(((size_t)blockIdx.x * blockDim.x + threadIdx.x) >> 5);
  int lane = threadIdx.x & 31;
  if (wid >= E) return;
  int s = src[wid], d = dst[wid];
  int h = lane >> 3;                 // head owning this lane's 16 columns
  float alpha = elog[(size_t)wid * HEADS + h] /
                (denom[(size_t)d * HEADS + h] + 1e-16f);
  const float4* vp = (const float4*)(v + (size_t)s * HID);
  float* mp = msg + (size_t)d * HID;
  #pragma unroll
  for (int i = 0; i < 4; ++i) {
    float4 val = vp[lane * 4 + i];
    int base = lane * 16 + i * 4;
    atomicAdd(&mp[base + 0], alpha * val.x);
    atomicAdd(&mp[base + 1], alpha * val.y);
    atomicAdd(&mp[base + 2], alpha * val.z);
    atomicAdd(&mp[base + 3], alpha * val.w);
  }
}

// ---------------------------------------------------------------------------
// h = LayerNorm(h + msg + skip) * g + b   (wave per row, in place)
// ---------------------------------------------------------------------------
__global__ __launch_bounds__(256) void ln_residual(
    float* __restrict__ h, const float* __restrict__ msg,
    const float* __restrict__ skp, const float* __restrict__ g,
    const float* __restrict__ b, int M)
{
  int row  = (int)(((size_t)blockIdx.x * blockDim.x + threadIdx.x) >> 5);
  int lane = threadIdx.x & 31;
  if (row >= M) return;
  const float4* hp = (const float4*)(h   + (size_t)row * HID);
  const float4* mp = (const float4*)(msg + (size_t)row * HID);
  const float4* sp = (const float4*)(skp + (size_t)row * HID);

  float vals[16];
  float sum = 0.0f;
  #pragma unroll
  for (int i = 0; i < 4; ++i) {
    float4 a = hp[lane * 4 + i], c = mp[lane * 4 + i], d = sp[lane * 4 + i];
    float x0 = a.x + c.x + d.x, x1 = a.y + c.y + d.y;
    float x2 = a.z + c.z + d.z, x3 = a.w + c.w + d.w;
    vals[i*4+0] = x0; vals[i*4+1] = x1; vals[i*4+2] = x2; vals[i*4+3] = x3;
    sum += x0 + x1 + x2 + x3;
  }
  #pragma unroll
  for (int m = 1; m < 32; m <<= 1) sum += __shfl_xor(sum, m, 32);
  float mean = sum * (1.0f / HID);

  float ss = 0.0f;
  #pragma unroll
  for (int j = 0; j < 16; ++j) { float dv = vals[j] - mean; ss += dv * dv; }
  #pragma unroll
  for (int m = 1; m < 32; m <<= 1) ss += __shfl_xor(ss, m, 32);
  float rstd = rsqrtf(ss * (1.0f / HID) + 1e-5f);

  float4* hw = (float4*)(h + (size_t)row * HID);
  const float4* gp = (const float4*)g;
  const float4* bp = (const float4*)b;
  #pragma unroll
  for (int i = 0; i < 4; ++i) {
    float4 gv = gp[lane * 4 + i], bv = bp[lane * 4 + i];
    float4 o;
    o.x = (vals[i*4+0] - mean) * rstd * gv.x + bv.x;
    o.y = (vals[i*4+1] - mean) * rstd * gv.y + bv.y;
    o.z = (vals[i*4+2] - mean) * rstd * gv.z + bv.z;
    o.w = (vals[i*4+3] - mean) * rstd * gv.w + bv.w;
    hw[lane * 4 + i] = o;
  }
}

// ---------------------------------------------------------------------------
static inline int cvtBlocks(size_t n) { return (int)((n / 8 + 255) / 256) + 1; }

extern "C" void kernel_launch(void* const* d_in, const int* in_sizes, int n_in,
                              void* d_out, int out_size, void* d_ws, size_t ws_size,
                              hipStream_t stream)
{
  (void)in_sizes; (void)n_in; (void)out_size; (void)ws_size;

  const float* emb  = (const float*)d_in[0];
  const int*   ei   = (const int*)  d_in[1];
  const float* inW  = (const float*)d_in[2];
  const float* inB  = (const float*)d_in[3];
  const float* Wq   = (const float*)d_in[4];
  const float* Bq   = (const float*)d_in[5];
  const float* Wk   = (const float*)d_in[6];
  const float* Bk   = (const float*)d_in[7];
  const float* Wv   = (const float*)d_in[8];
  const float* Bv   = (const float*)d_in[9];
  const float* Ws   = (const float*)d_in[10];
  const float* Bs_  = (const float*)d_in[11];
  const float* lnG  = (const float*)d_in[12];
  const float* lnB  = (const float*)d_in[13];
  const float* outW = (const float*)d_in[14];
  const float* outB = (const float*)d_in[15];
  float* out = (float*)d_out;

  const int* srcp = ei;             // edge_index[0]
  const int* dstp = ei + N_EDGES;   // edge_index[1]

  // ---- workspace carve-up: f32 region, then 16B-aligned f16 region ----
  float* ws = (float*)d_ws;
  const size_t NH = (size_t)N_NODES * HID;
  float* h     = ws;
  float* q     = h + NH;
  float* k     = q + NH;
  float* v     = k + NH;
  float* skp   = v + NH;
  float* msg   = skp + NH;
  float* elog  = msg + NH;                      // E*HEADS
  float* nmax  = elog + (size_t)N_EDGES * HEADS;
  float* denom = nmax + (size_t)N_NODES * HEADS;

  _Float16* h16    = (_Float16*)(denom + (size_t)N_NODES * HEADS);
  _Float16* emb16  = h16   + NH;
  _Float16* inW16  = emb16 + (size_t)N_NODES * IN_DIM;
  _Float16* wq16   = inW16 + (size_t)IN_DIM * HID;
  _Float16* wk16   = wq16  + (size_t)LAYERS * HID * HID;
  _Float16* wv16   = wk16  + (size_t)LAYERS * HID * HID;
  _Float16* wsk16  = wv16  + (size_t)LAYERS * HID * HID;
  _Float16* outW16 = wsk16 + (size_t)LAYERS * HID * HID;

  dim3 blk(256);

  // ---- one-time f16 conversions ----
  cvt_f32_f16<<<cvtBlocks((size_t)N_NODES * IN_DIM), blk, 0, stream>>>(
      emb, emb16, (size_t)N_NODES * IN_DIM);
  cvt_f32_f16<<<cvtBlocks((size_t)IN_DIM * HID), blk, 0, stream>>>(
      inW, inW16, (size_t)IN_DIM * HID);
  cvt_f32_f16<<<cvtBlocks((size_t)LAYERS * HID * HID), blk, 0, stream>>>(
      Wq, wq16, (size_t)LAYERS * HID * HID);
  cvt_f32_f16<<<cvtBlocks((size_t)LAYERS * HID * HID), blk, 0, stream>>>(
      Wk, wk16, (size_t)LAYERS * HID * HID);
  cvt_f32_f16<<<cvtBlocks((size_t)LAYERS * HID * HID), blk, 0, stream>>>(
      Wv, wv16, (size_t)LAYERS * HID * HID);
  cvt_f32_f16<<<cvtBlocks((size_t)LAYERS * HID * HID), blk, 0, stream>>>(
      Ws, wsk16, (size_t)LAYERS * HID * HID);
  cvt_f32_f16<<<cvtBlocks((size_t)OUT_DIM * HID), blk, 0, stream>>>(
      outW, outW16, (size_t)OUT_DIM * HID);

  // input projection: h = emb @ inW^T + inB   [50000,386] -> [50000,512]
  launch_gemm(emb16, inW16, inB, h, N_NODES, HID, IN_DIM, stream);

  const int fillBlocks = (int)((NH + 255) / 256);
  const int waveBlocks = (N_EDGES * 32 + 255) / 256;           // wave per edge
  const int ehBlocks   = (N_EDGES * HEADS + 255) / 256;
  const int lnBlocks   = (N_NODES + 7) / 8;                    // 8 waves/block
  const int hcBlocks   = cvtBlocks(NH);

  for (int l = 0; l < LAYERS; ++l) {
    const float* bq  = Bq  + (size_t)l * HID;
    const float* bk  = Bk  + (size_t)l * HID;
    const float* bv  = Bv  + (size_t)l * HID;
    const float* bsk = Bs_ + (size_t)l * HID;
    const _Float16* wq  = wq16  + (size_t)l * HID * HID;
    const _Float16* wk  = wk16  + (size_t)l * HID * HID;
    const _Float16* wv  = wv16  + (size_t)l * HID * HID;
    const _Float16* wsk = wsk16 + (size_t)l * HID * HID;

    cvt_f32_f16<<<hcBlocks, blk, 0, stream>>>(h, h16, NH);

    launch_gemm(h16, wq,  bq,  q,   N_NODES, HID, HID, stream);
    launch_gemm(h16, wk,  bk,  k,   N_NODES, HID, HID, stream);
    launch_gemm(h16, wv,  bv,  v,   N_NODES, HID, HID, stream);
    launch_gemm(h16, wsk, bsk, skp, N_NODES, HID, HID, stream);

    fill_layer<<<fillBlocks, blk, 0, stream>>>(msg, nmax, denom);
    edge_logits<<<waveBlocks, blk, 0, stream>>>(q, k, srcp, dstp, elog, nmax,
                                                N_EDGES);
    edge_exp<<<ehBlocks, blk, 0, stream>>>(elog, nmax, denom, dstp, N_EDGES);
    edge_scatter<<<waveBlocks, blk, 0, stream>>>(v, elog, denom, srcp, dstp,
                                                 msg, N_EDGES);
    ln_residual<<<lnBlocks, blk, 0, stream>>>(h, msg, skp,
                                              lnG + (size_t)l * HID,
                                              lnB + (size_t)l * HID, N_NODES);
  }

  // output projection: out = h @ outW^T + outB   [50000,512] -> [50000,384]
  cvt_f32_f16<<<hcBlocks, blk, 0, stream>>>(h, h16, NH);
  launch_gemm(h16, outW16, outB, out, N_NODES, OUT_DIM, HID, stream);
}
